// GraphEncoderDGL_3049426780782
// MI455X (gfx1250) — compile-verified, matrix operand order checked by reading, not run on hardware
//
#include <hip/hip_runtime.h>
#include <hip/hip_bf16.h>

// ---------- types for CDNA5 WMMA ----------
typedef __attribute__((ext_vector_type(16))) __bf16        v16bf;
typedef __attribute__((ext_vector_type(8)))  float         v8f;
typedef __attribute__((ext_vector_type(8)))  unsigned short ushort8;

union Frag { ushort8 h[2]; v16bf v; };

// ---------- bf16 helpers (stored as ushort) ----------
__device__ __forceinline__ unsigned short f2bf(float f) {
  unsigned u = __float_as_uint(f);
  u += 0x7FFFu + ((u >> 16) & 1u);           // round-to-nearest-even
  return (unsigned short)(u >> 16);
}
__device__ __forceinline__ float bf2f(unsigned short h) {
  return __uint_as_float((unsigned)h << 16);
}

// ---------- zero small scratch ----------
__global__ void zero_kernel(float* p, int n) {
  int i = blockIdx.x * blockDim.x + threadIdx.x;
  if (i < n) p[i] = 0.f;
}

// ---------- stem: 1x1 conv (3->64), raw bf16 out ----------
__global__ __launch_bounds__(256)
void stem_kernel(const float* __restrict__ x, const float* __restrict__ w,
                 unsigned short* __restrict__ h0, int N) {
  __shared__ float sw[192];
  int tid = threadIdx.x;
  if (tid < 192) sw[tid] = w[tid];
  __syncthreads();
  int b = blockIdx.y;
  int n = blockIdx.x * 256 + tid;
  size_t xb = (size_t)b * 3 * N;
  float x0 = x[xb + n], x1 = x[xb + N + n], x2 = x[xb + 2 * (size_t)N + n];
  size_t ob = (size_t)b * 64 * N;
  #pragma unroll 8
  for (int c = 0; c < 64; ++c) {
    float v = sw[c * 3] * x0 + sw[c * 3 + 1] * x1 + sw[c * 3 + 2] * x2;
    h0[ob + (size_t)c * N + n] = f2bf(v);
  }
}

// ---------- per-channel sum / sumsq over (B,N) ----------
__global__ __launch_bounds__(256)
void stats_kernel(const unsigned short* __restrict__ h, int C, int N,
                  float* __restrict__ sum, float* __restrict__ sumsq) {
  int c = blockIdx.x, b = blockIdx.y, tid = threadIdx.x;
  const unsigned short* p = h + ((size_t)b * C + c) * N;
  float s = 0.f, q = 0.f;
  for (int n = tid; n < N; n += 256) { float v = bf2f(p[n]); s += v; q += v * v; }
  __shared__ float rs[256], rq[256];
  rs[tid] = s; rq[tid] = q;
  __syncthreads();
  for (int st = 128; st > 0; st >>= 1) {
    if (tid < st) { rs[tid] += rs[tid + st]; rq[tid] += rq[tid + st]; }
    __syncthreads();
  }
  if (tid == 0) { atomicAdd(&sum[c], rs[0]); atomicAdd(&sumsq[c], rq[0]); }
}

// ---------- fold BN into scale/shift ----------
__global__ void finalize_kernel(const float* sum, const float* sumsq,
                                const float* g, const float* be,
                                float* scale, float* shift, int C, float inv) {
  int c = blockIdx.x * blockDim.x + threadIdx.x;
  if (c < C) {
    float m = sum[c] * inv;
    float v = sumsq[c] * inv - m * m;
    float r = rsqrtf(v + 1e-5f);
    float sc = g[c] * r;
    scale[c] = sc;
    shift[c] = be[c] - m * sc;
  }
}

// ---------- implicit-GEMM Conv1d k=3 s=2 pad=1 via bf16 WMMA ----------
// block: 256 threads = 8 waves; tile 64 Cout x 128 out positions (one batch b).
// wave grid 4x2: wm in {0..3} (16 Cout rows each), wn in {0,1} (64 cols = four
// 16x16 N tiles each -> 4 accumulators, 4 wmma per K-step per wave).
// act applied to INPUT: 0 = LeakyReLU(0.2) (stem output), 1 = ReLU.
__global__ __launch_bounds__(256)
void ds_conv_wmma(const unsigned short* __restrict__ in,
                  const float* __restrict__ w,
                  const float* __restrict__ scale,
                  const float* __restrict__ shift,
                  unsigned short* __restrict__ out,
                  int Cin, int Nin, int Cout, int act) {
  const int Nout = Nin >> 1;
  const int j0  = blockIdx.x * 128;
  const int co0 = blockIdx.y * 64;
  const int b   = blockIdx.z;
  const int tid = threadIdx.x;
  const int lane = tid & 31;
  const int wid  = tid >> 5;
  const int wm = wid & 3, wn = wid >> 2;

  __shared__ unsigned short As[64][40];    // [m][k], 80B rows (16B-aligned frags)
  __shared__ unsigned short Bs[128][40];   // [j][k] (transposed: row-contiguous frags)
  __shared__ float sS[256], sH[256];       // folded BN scale/shift (Cin <= 256)

  for (int c = tid; c < Cin; c += 256) { sS[c] = scale[c]; sH[c] = shift[c]; }

  const float leak = (act == 0) ? 0.2f : 0.f;

  v8f acc[4];
  #pragma unroll
  for (int s = 0; s < 4; ++s) acc[s] = (v8f){0.f,0.f,0.f,0.f,0.f,0.f,0.f,0.f};

  const size_t inBase = (size_t)b * Cin * Nin;
  const int mrow = wm * 16 + (lane & 15);
  const int nbase = wn * 64 + (lane & 15);
  const int K0   = (lane >> 4) << 3;       // lanes 0-15: K0=0, lanes 16-31: K0=8

  for (int t = 0; t < 3; ++t) {
    for (int kc = 0; kc < Cin; kc += 32) {
      __syncthreads();
      // stage A: weights tile (64 Cout x 32 K), f32->bf16
      #pragma unroll
      for (int i = tid; i < 64 * 32; i += 256) {
        int m = i >> 5, kk = i & 31;
        As[m][kk] = f2bf(w[(size_t)(co0 + m) * Cin * 3 + (size_t)(kc + kk) * 3 + t]);
      }
      // stage B: normalized+activated inputs (128 pos x 32 K), zero-pad n=-1
      #pragma unroll
      for (int i = tid; i < 128 * 32; i += 256) {
        int j = i & 127, kk = i >> 7;
        int ci = kc + kk;
        int n = 2 * (j0 + j) - 1 + t;
        float v = 0.f;
        if ((unsigned)n < (unsigned)Nin) {
          float r = bf2f(in[inBase + (size_t)ci * Nin + n]);
          r = r * sS[ci] + sH[ci];
          v = (r > 0.f) ? r : leak * r;
        }
        Bs[j][kk] = f2bf(v);
      }
      __syncthreads();

      // per-lane fragments per ISA 16-bit A/B striping: K = {K0..K0+7, K0+16..K0+23}
      Frag fa;
      fa.h[0] = *(const ushort8*)&As[mrow][K0];
      fa.h[1] = *(const ushort8*)&As[mrow][K0 + 16];
      #pragma unroll
      for (int s = 0; s < 4; ++s) {
        Frag fb;
        fb.h[0] = *(const ushort8*)&Bs[nbase + s * 16][K0];
        fb.h[1] = *(const ushort8*)&Bs[nbase + s * 16][K0 + 16];
        acc[s] = __builtin_amdgcn_wmma_f32_16x16x32_bf16(false, fa.v, false, fb.v,
                                                         (short)0, acc[s], false, false);
      }
    }
  }

  // epilogue: C layout — VGPR r: lanes 0-15 M=r, lanes 16-31 M=r+8; N = lane%16.
  // ds-stage biases are dropped: training-mode BN cancels per-channel constants.
  const int mlo  = (lane >> 4) << 3;
  const int ncol = lane & 15;
  const size_t outBase = (size_t)b * Cout * Nout;
  #pragma unroll
  for (int s = 0; s < 4; ++s) {
    int j = j0 + wn * 64 + s * 16 + ncol;
    #pragma unroll
    for (int r = 0; r < 8; ++r) {
      int co = co0 + wm * 16 + mlo + r;
      out[outBase + (size_t)co * Nout + j] = f2bf(acc[s][r]);
    }
  }
}

// ---------- mean over nodes of relu(bn(h3)) -> hbar (B,512) ----------
__global__ __launch_bounds__(256)
void meanpool_kernel(const unsigned short* __restrict__ h,
                     const float* __restrict__ scale, const float* __restrict__ shift,
                     float* __restrict__ hbar, int C, int N) {
  int c = blockIdx.x, b = blockIdx.y, tid = threadIdx.x;
  const unsigned short* p = h + ((size_t)b * C + c) * N;
  float sc = scale[c], sh = shift[c];
  float s = 0.f;
  for (int n = tid; n < N; n += 256) {
    float v = bf2f(p[n]) * sc + sh;
    s += (v > 0.f) ? v : 0.f;
  }
  __shared__ float rs[256];
  rs[tid] = s; __syncthreads();
  for (int st = 128; st > 0; st >>= 1) {
    if (tid < st) rs[tid] += rs[tid + st];
    __syncthreads();
  }
  if (tid == 0) hbar[(size_t)b * C + c] = rs[0] / (float)N;
}

// ---------- proj: out[b,o] = proj_w[o,:]·hbar[b,:] + proj_b[o] ----------
__global__ __launch_bounds__(256)
void proj_kernel(const float* __restrict__ hbar, const float* __restrict__ w,
                 const float* __restrict__ bias, float* __restrict__ out) {
  int b = blockIdx.y;
  int o = blockIdx.x * 256 + threadIdx.x;   // grid.x = 4 -> o in [0,1024)
  __shared__ float hb[512];
  for (int c = threadIdx.x; c < 512; c += 256) hb[c] = hbar[(size_t)b * 512 + c];
  __syncthreads();
  const float* wr = w + (size_t)o * 512;
  float acc = bias[o];
  #pragma unroll 4
  for (int c = 0; c < 512; ++c) acc += wr[c] * hb[c];
  out[(size_t)b * 1024 + o] = acc;
}

extern "C" void kernel_launch(void* const* d_in, const int* in_sizes, int n_in,
                              void* d_out, int out_size, void* d_ws, size_t ws_size,
                              hipStream_t stream) {
  (void)in_sizes; (void)n_in; (void)out_size; (void)ws_size;
  const float* x      = (const float*)d_in[0];
  const float* stem_w = (const float*)d_in[1];
  const float* stem_g = (const float*)d_in[2];
  const float* stem_b = (const float*)d_in[3];
  const float* ds_w[3]  = {(const float*)d_in[4],  (const float*)d_in[8],  (const float*)d_in[12]};
  const float* ds_g[3]  = {(const float*)d_in[6],  (const float*)d_in[10], (const float*)d_in[14]};
  const float* ds_be[3] = {(const float*)d_in[7],  (const float*)d_in[11], (const float*)d_in[15]};
  const float* proj_w = (const float*)d_in[16];
  const float* proj_b = (const float*)d_in[17];
  float* out = (float*)d_out;

  const int B = 64, N0 = 8192;
  const size_t BUF = 67108864;  // 33.5M bf16 elements (every stage is the same size)
  char* wsb = (char*)d_ws;
  unsigned short* bufA = (unsigned short*)wsb;
  unsigned short* bufB = (unsigned short*)(wsb + BUF);
  float* stats = (float*)(wsb + 2 * BUF);  // 1920 floats: per-stage sum || sumsq
  float* ssc   = stats + 1920;             // 1920 floats: per-stage scale || shift
  float* hbar  = ssc + 1920;               // 64*512 floats

  const int soff[4] = {0, 128, 384, 896};
  const int Cs[4]   = {64, 128, 256, 512};
  const int Ns[4]   = {8192, 4096, 2048, 1024};
  unsigned short* bufs[4] = {bufA, bufB, bufA, bufB};
  const float* gs[4]  = {stem_g, ds_g[0], ds_g[1], ds_g[2]};
  const float* bes[4] = {stem_b, ds_be[0], ds_be[1], ds_be[2]};

  zero_kernel<<<(1920 + 255) / 256, 256, 0, stream>>>(stats, 1920);
  stem_kernel<<<dim3(N0 / 256, B), 256, 0, stream>>>(x, stem_w, bufA, N0);

  for (int s = 0; s < 4; ++s) {
    stats_kernel<<<dim3(Cs[s], B), 256, 0, stream>>>(
        bufs[s], Cs[s], Ns[s], stats + soff[s], stats + soff[s] + Cs[s]);
    finalize_kernel<<<(Cs[s] + 255) / 256, 256, 0, stream>>>(
        stats + soff[s], stats + soff[s] + Cs[s], gs[s], bes[s],
        ssc + soff[s], ssc + soff[s] + Cs[s], Cs[s], 1.0f / (float)(B * Ns[s]));
    if (s < 3) {
      ds_conv_wmma<<<dim3(Ns[s + 1] / 128, Cs[s + 1] / 64, B), 256, 0, stream>>>(
          bufs[s], ds_w[s], ssc + soff[s], ssc + soff[s] + Cs[s],
          bufs[s + 1], Cs[s], Ns[s], Cs[s + 1], (s == 0) ? 0 : 1);
    }
  }

  meanpool_kernel<<<dim3(512, B), 256, 0, stream>>>(
      bufB, ssc + soff[3], ssc + soff[3] + 512, hbar, 512, 1024);
  proj_kernel<<<dim3(4, B), 256, 0, stream>>>(hbar, proj_w, proj_b, out);
}
